// MambaSimple_67602785239495
// MI455X (gfx1250) — matrix operand order repaired and checked by reading, not compiled, not run on hardware
//
#include <hip/hip_runtime.h>
#include <hip/hip_bf16.h>

// ---------------------------------------------------------------------------
// Mamba block forward for MI455X (gfx1250, wave32, WMMA bf16 -> f32 accum)
// ---------------------------------------------------------------------------

typedef __bf16 bf16_t;
typedef __attribute__((ext_vector_type(16))) __bf16 v16bf;
typedef __attribute__((ext_vector_type(8)))  __bf16 v8bf;
typedef __attribute__((ext_vector_type(8)))  float   v8f;
typedef int v4i_vs __attribute__((vector_size(16)));   // matches builtin param

#define D_MODEL 1024
#define D_STATE 16
#define D_CONV  4
#define D_INNER 2048
#define BATCH   2
#define SEQ     1024
#define BL      (BATCH * SEQ)           // 2048 tokens
#define XDBL_W  (D_MODEL + 2 * D_STATE) // 1056

// Async global->LDS (gfx1250 ASYNCcnt path), guarded: falls back to the
// synchronous staging path if the builtin is not available.
#if defined(__has_builtin)
#  if __has_builtin(__builtin_amdgcn_global_load_async_to_lds_b128)
#    define HAVE_ASYNC_LDS 1
#  endif
#endif
#ifndef HAVE_ASYNC_LDS
#  define HAVE_ASYNC_LDS 0
#endif

#if HAVE_ASYNC_LDS
#  if __has_builtin(__builtin_amdgcn_s_wait_asynccnt)
#    define WAIT_ASYNC0() __builtin_amdgcn_s_wait_asynccnt(0)
#  else
#    define WAIT_ASYNC0() asm volatile("s_wait_asynccnt 0" ::: "memory")
#  endif

__device__ __forceinline__ void async_copy_b128(const float* src, float* dst) {
    __builtin_amdgcn_global_load_async_to_lds_b128(
        (__attribute__((address_space(1))) v4i_vs*)(void*)src,
        (__attribute__((address_space(3))) v4i_vs*)(void*)dst,
        0, 0);
}
#endif

__device__ __forceinline__ float silu_f(float v) {
    return v / (1.f + __expf(-v));
}
__device__ __forceinline__ float softplus_f(float v) {
    return (v > 20.f) ? v : log1pf(__expf(v));
}

// ---------------------------------------------------------------------------
// Generic bf16 WMMA GEMM:  C[M x N] (f32, row-major) = A[M x K] (bf16, row-major)
//                          x Bt[N x K] (bf16, B pre-transposed to N-major)
// One wave per (16*MT x 16*NT) output tile, register-blocked MT x NT.
// M % (16*MT) == 0, N % (16*NT) == 0, K % 32 == 0.
// ---------------------------------------------------------------------------
template <int MT, int NT>
__global__ void __launch_bounds__(32)
gemm_bf16_wmma(const bf16_t* __restrict__ A, const bf16_t* __restrict__ Bt,
               float* __restrict__ C, int M, int N, int K)
{
    const int lane = threadIdx.x & 31;
    const int half = lane >> 4;        // 0: lanes 0-15, 1: lanes 16-31
    const int lq   = lane & 15;
    const int n0   = blockIdx.x * (16 * NT);
    const int m0   = blockIdx.y * (16 * MT);

    v8f acc[MT][NT];
    #pragma unroll
    for (int i = 0; i < MT; ++i)
        #pragma unroll
        for (int j = 0; j < NT; ++j)
            acc[i][j] = v8f{};

    // A fragment base: lane lq = row, two 8-elem (16B) chunks at K offsets
    // half*8 and 16+half*8 (ISA 16-bit A 16x32 layout).
    const bf16_t* ap[MT];
    #pragma unroll
    for (int i = 0; i < MT; ++i)
        ap[i] = A + (size_t)(m0 + 16 * i + lq) * K + half * 8;
    // B fragment base from N-major Bt: lane lq = col N, two 8-elem chunks at K
    // offsets half*8 and 16+half*8 (matching the A fragment K order).
    const bf16_t* bp[NT];
    #pragma unroll
    for (int j = 0; j < NT; ++j)
        bp[j] = Bt + (size_t)(n0 + 16 * j + lq) * K + half * 8;

    for (int k0 = 0; k0 < K; k0 += 32) {
        __builtin_prefetch(ap[0] + k0 + 64, 0, 1);  // -> global_prefetch_b8
        __builtin_prefetch(bp[0] + k0 + 64, 0, 1);

        v16bf af[MT], bfv[NT];
        #pragma unroll
        for (int i = 0; i < MT; ++i) {
            v8bf lo = *(const v8bf*)(ap[i] + k0);
            v8bf hi = *(const v8bf*)(ap[i] + k0 + 16);
            af[i] = __builtin_shufflevector(lo, hi,
                        0,1,2,3,4,5,6,7,8,9,10,11,12,13,14,15);
        }
        #pragma unroll
        for (int j = 0; j < NT; ++j) {
            v8bf lo = *(const v8bf*)(bp[j] + k0);
            v8bf hi = *(const v8bf*)(bp[j] + k0 + 16);
            bfv[j] = __builtin_shufflevector(lo, hi,
                        0,1,2,3,4,5,6,7,8,9,10,11,12,13,14,15);
        }
        #pragma unroll
        for (int i = 0; i < MT; ++i)
            #pragma unroll
            for (int j = 0; j < NT; ++j)
                acc[i][j] = __builtin_amdgcn_wmma_f32_16x16x32_bf16(
                    false, af[i], false, bfv[j], (short)0, acc[i][j],
                    false, false);
    }

    // C/D layout: col = lane%16, row = half*8 + r (r = VGPR index 0..7)
    #pragma unroll
    for (int i = 0; i < MT; ++i) {
        const int rbase = m0 + 16 * i + half * 8;
        #pragma unroll
        for (int j = 0; j < NT; ++j) {
            const int col = n0 + 16 * j + lq;
            #pragma unroll
            for (int r = 0; r < 8; ++r)
                C[(size_t)(rbase + r) * N + col] = acc[i][j][r];
        }
    }
}

// ---------------------------------------------------------------------------
// f32 -> bf16, row-major copy with independent row strides (repack)
// ---------------------------------------------------------------------------
__global__ void __launch_bounds__(256)
convert_bf16_strided(const float* __restrict__ in, bf16_t* __restrict__ out,
                     int rows, int cols, int in_stride, int out_stride)
{
    int idx = blockIdx.x * 256 + threadIdx.x;
    if (idx >= rows * cols) return;
    int r = idx / cols, c = idx % cols;
    out[(size_t)r * out_stride + c] = (bf16_t)in[(size_t)r * in_stride + c];
}

// ---------------------------------------------------------------------------
// f32 [rows x cols] -> bf16 transposed [cols x rows]  (weights -> N-major)
// ---------------------------------------------------------------------------
__global__ void __launch_bounds__(256)
convert_bf16_transpose(const float* __restrict__ in, bf16_t* __restrict__ out,
                       int rows, int cols)
{
    int idx = blockIdx.x * 256 + threadIdx.x;
    if (idx >= rows * cols) return;
    int r = idx / cols, c = idx % cols;
    out[(size_t)c * rows + r] = (bf16_t)in[(size_t)r * cols + c];
}

// ---------------------------------------------------------------------------
// Causal depthwise conv1d (width 4) + bias + SiLU over the x half of xz.
// xz: [BL x 4096]; writes x as f32 [BL x 2048] and bf16 (for next GEMM).
// ---------------------------------------------------------------------------
__global__ void __launch_bounds__(256)
conv_silu_kernel(const float* __restrict__ xz, const float* __restrict__ conv_w,
                 const float* __restrict__ conv_b, float* __restrict__ xf,
                 bf16_t* __restrict__ xb)
{
    int gid = blockIdx.x * 256 + threadIdx.x;
    if (gid >= BL * D_INNER) return;
    int c   = gid % D_INNER;
    int row = gid / D_INNER;     // b*SEQ + l
    int l   = row % SEQ;

    float acc = conv_b[c];
    #pragma unroll
    for (int w = 0; w < D_CONV; ++w) {
        int li = l - (D_CONV - 1) + w;
        if (li >= 0)
            acc += conv_w[w * D_INNER + c] *
                   xz[(size_t)(row - (D_CONV - 1) + w) * (2 * D_INNER) + c];
    }
    float s = silu_f(acc);
    xf[gid] = s;
    xb[gid] = (bf16_t)s;
}

// ---------------------------------------------------------------------------
// Selective scan. One thread per (batch, inner-channel); 16 states in VGPRs.
// B/C (shared per batch per step) staged through LDS in 16-step chunks.
// With async-to-LDS support: double-buffered, wave 0 DMAs chunk n+1 while
// all waves compute chunk n (overlap on the latency-bound serial stage).
// y = sum_s state*C + u  (D == 1); dt = softplus(dt_pre + b_dt) fused.
// ---------------------------------------------------------------------------
__global__ void __launch_bounds__(256)
scan_kernel(const float* __restrict__ x_dbl,  // [BL x 1056]
            const float* __restrict__ dt_pre, // [BL x 2048]
            const float* __restrict__ b_dt,   // [2048]
            const float* __restrict__ A_log,  // [2048 x 16]
            const float* __restrict__ u,      // [BL x 2048]
            float* __restrict__ y)            // [BL x 2048]
{
    __shared__ float sBC[2][16 * 32];         // 2 x (16 steps x (16 B + 16 C))
    const int tid = threadIdx.x;
    const int gid = blockIdx.x * 256 + tid;
    const int b   = gid / D_INNER;            // uniform per block (2048%256==0)
    const int i   = gid % D_INNER;

    float Arow[D_STATE];
    #pragma unroll
    for (int s = 0; s < D_STATE; ++s) Arow[s] = -__expf(A_log[i * D_STATE + s]);
    float st[D_STATE];
    #pragma unroll
    for (int s = 0; s < D_STATE; ++s) st[s] = 0.f;
    const float bdt = b_dt[i];

#if HAVE_ASYNC_LDS
    // Wave 0 copies one 2048B chunk (16 rows x 128B of B|C) with 4 async
    // b128 ops: lane covers row (l0 + 4q + tid/8), 16B slice (tid%8)*16.
    auto issue_chunk = [&](int l0, int nbuf) {
        const int stp  = tid >> 3;            // 0..3
        const int col4 = (tid & 7) * 4;       // float offset in row
        #pragma unroll
        for (int q = 0; q < 4; ++q) {
            const int step = q * 4 + stp;
            const float* src = x_dbl +
                (size_t)(b * SEQ + l0 + step) * XDBL_W + D_MODEL + col4;
            async_copy_b128(src, &sBC[nbuf][step * 32 + col4]);
        }
    };

    if (tid < 32) { issue_chunk(0, 0); WAIT_ASYNC0(); }
    __syncthreads();

    for (int ci = 0; ci < SEQ / 16; ++ci) {
        const int l0  = ci * 16;
        const int buf = ci & 1;
        if (tid < 32 && ci + 1 < SEQ / 16) issue_chunk(l0 + 16, buf ^ 1);
#else
    for (int ci = 0; ci < SEQ / 16; ++ci) {
        const int l0  = ci * 16;
        const int buf = 0;
        __syncthreads();
        for (int f = tid; f < 512; f += 256) {
            int step = f >> 5, j = f & 31;
            sBC[0][f] =
                x_dbl[(size_t)(b * SEQ + l0 + step) * XDBL_W + D_MODEL + j];
        }
        __syncthreads();
#endif
        for (int t = 0; t < 16; ++t) {
            const size_t row = (size_t)(b * SEQ + l0 + t);
            float dv = softplus_f(dt_pre[row * D_INNER + i] + bdt);
            const float uv = u[row * D_INNER + i];
            float yv = uv;                    // D = 1
            const float* bc = &sBC[buf][t * 32];
            #pragma unroll
            for (int s = 0; s < D_STATE; ++s) {
                st[s] = __expf(dv * Arow[s]) * st[s] + (dv * bc[s]) * uv;
                yv += st[s] * bc[16 + s];
            }
            y[row * D_INNER + i] = yv;
        }
#if HAVE_ASYNC_LDS
        if (tid < 32) WAIT_ASYNC0();
        __syncthreads();
#endif
    }
}

// ---------------------------------------------------------------------------
// y * silu(z)  -> bf16 for the output GEMM.  z = xz[:, 2048:4096].
// ---------------------------------------------------------------------------
__global__ void __launch_bounds__(256)
gate_kernel(const float* __restrict__ y, const float* __restrict__ xz,
            bf16_t* __restrict__ ygate)
{
    int gid = blockIdx.x * 256 + threadIdx.x;
    if (gid >= BL * D_INNER) return;
    int row = gid / D_INNER, c = gid % D_INNER;
    float z = xz[(size_t)row * (2 * D_INNER) + D_INNER + c];
    ygate[gid] = (bf16_t)(y[gid] * silu_f(z));
}

// ---------------------------------------------------------------------------
// Host-side launch
// ---------------------------------------------------------------------------
static inline char* carve(char*& p, size_t bytes) {
    char* r = p;
    p += (bytes + 255) & ~(size_t)255;
    return r;
}

extern "C" void kernel_launch(void* const* d_in, const int* in_sizes, int n_in,
                              void* d_out, int out_size, void* d_ws, size_t ws_size,
                              hipStream_t stream) {
    (void)in_sizes; (void)n_in; (void)out_size; (void)ws_size;

    const float* inputs = (const float*)d_in[0];  // [BL x 1024]
    const float* W_in   = (const float*)d_in[1];  // [1024 x 4096]
    const float* conv_w = (const float*)d_in[2];  // [4 x 1 x 2048]
    const float* conv_b = (const float*)d_in[3];  // [2048]
    const float* W_x    = (const float*)d_in[4];  // [2048 x 1056]
    const float* W_dt   = (const float*)d_in[5];  // [1024 x 2048]
    const float* b_dt   = (const float*)d_in[6];  // [2048]
    const float* A_log  = (const float*)d_in[7];  // [2048 x 16]
    const float* W_out  = (const float*)d_in[8];  // [2048 x 1024]
    float* out = (float*)d_out;                   // [BL x 1024]

    char* p = (char*)d_ws;
    float*  xz      = (float*) carve(p, (size_t)BL * 4096 * 4);
    float*  x_f32   = (float*) carve(p, (size_t)BL * D_INNER * 4);
    bf16_t* x_bf    = (bf16_t*)carve(p, (size_t)BL * D_INNER * 2);
    float*  x_dbl   = (float*) carve(p, (size_t)BL * XDBL_W * 4);
    float*  dt_pre  = (float*) carve(p, (size_t)BL * D_INNER * 4);
    bf16_t* dtin_bf = (bf16_t*)carve(p, (size_t)BL * D_MODEL * 2);
    float*  y_scan  = (float*) carve(p, (size_t)BL * D_INNER * 4);
    bf16_t* ygate   = (bf16_t*)carve(p, (size_t)BL * D_INNER * 2);
    bf16_t* in_bf   = (bf16_t*)carve(p, (size_t)BL * D_MODEL * 2);
    bf16_t* WinT    = (bf16_t*)carve(p, (size_t)4096 * 1024 * 2);
    bf16_t* WxT     = (bf16_t*)carve(p, (size_t)XDBL_W * 2048 * 2);
    bf16_t* WdtT    = (bf16_t*)carve(p, (size_t)2048 * 1024 * 2);
    bf16_t* WoutT   = (bf16_t*)carve(p, (size_t)1024 * 2048 * 2);

    const int TB = 256;
    auto blocks = [](size_t n, int tb) { return (unsigned)((n + tb - 1) / tb); };

    // Precision conversions / weight transposes (B operands -> N-major bf16)
    convert_bf16_strided<<<blocks((size_t)BL * D_MODEL, TB), TB, 0, stream>>>(
        inputs, in_bf, BL, D_MODEL, D_MODEL, D_MODEL);
    convert_bf16_transpose<<<blocks((size_t)1024 * 4096, TB), TB, 0, stream>>>(
        W_in, WinT, 1024, 4096);
    convert_bf16_transpose<<<blocks((size_t)2048 * XDBL_W, TB), TB, 0, stream>>>(
        W_x, WxT, 2048, XDBL_W);
    convert_bf16_transpose<<<blocks((size_t)1024 * 2048, TB), TB, 0, stream>>>(
        W_dt, WdtT, 1024, 2048);
    convert_bf16_transpose<<<blocks((size_t)2048 * 1024, TB), TB, 0, stream>>>(
        W_out, WoutT, 2048, 1024);

    // GEMM1: xz = inputs @ W_in          [2048 x 4096], K=1024, 64x64 tiles
    gemm_bf16_wmma<4, 4><<<dim3(4096 / 64, BL / 64), 32, 0, stream>>>(
        in_bf, WinT, xz, BL, 4096, 1024);

    // Conv + bias + SiLU on x half
    conv_silu_kernel<<<blocks((size_t)BL * D_INNER, TB), TB, 0, stream>>>(
        xz, conv_w, conv_b, x_f32, x_bf);

    // GEMM2: x_dbl = x @ W_x             [2048 x 1056], K=2048, 64x32 tiles
    gemm_bf16_wmma<4, 2><<<dim3(XDBL_W / 32, BL / 64), 32, 0, stream>>>(
        x_bf, WxT, x_dbl, BL, XDBL_W, 2048);

    // dt_in (first 1024 cols of x_dbl) -> bf16
    convert_bf16_strided<<<blocks((size_t)BL * D_MODEL, TB), TB, 0, stream>>>(
        x_dbl, dtin_bf, BL, D_MODEL, XDBL_W, D_MODEL);

    // GEMM3: dt_pre = dt_in @ W_dt       [2048 x 2048], K=1024, 64x64 tiles
    gemm_bf16_wmma<4, 4><<<dim3(D_INNER / 64, BL / 64), 32, 0, stream>>>(
        dtin_bf, WdtT, dt_pre, BL, D_INNER, 1024);

    // Selective scan (4096 channel-threads, 16 blocks)
    scan_kernel<<<(BATCH * D_INNER) / TB, TB, 0, stream>>>(
        x_dbl, dt_pre, b_dt, A_log, x_f32, y_scan);

    // Gate: y * silu(z) -> bf16
    gate_kernel<<<blocks((size_t)BL * D_INNER, TB), TB, 0, stream>>>(
        y_scan, xz, ygate);

    // GEMM4: out = ygate @ W_out         [2048 x 1024], K=2048, 64x64 tiles
    gemm_bf16_wmma<4, 4><<<dim3(D_MODEL / 64, BL / 64), 32, 0, stream>>>(
        ygate, WoutT, out, BL, D_MODEL, 2048);
}
